// GCNClassifier_88648124990263
// MI455X (gfx1250) — matrix operand order repaired and checked by reading, not compiled, not run on hardware
//
#include <hip/hip_runtime.h>
#include <hip/hip_bf16.h>

typedef __attribute__((ext_vector_type(16))) __bf16          v16bf;
typedef __attribute__((ext_vector_type(8)))  float           v8f;
typedef __attribute__((ext_vector_type(16))) unsigned short  v16u;
typedef __attribute__((ext_vector_type(8)))  unsigned short  v8u;

#define DH   128      // emb_dim == hidden_dim
#define TM   64       // rows per GEMM block
#define LDP  (DH + 8) // padded LDS row: +16B staggers banks by 4/row -> conflict-free

static __device__ __forceinline__ unsigned short f2bf(float f) {
    unsigned int u = __float_as_uint(f);
    u += 0x7FFFu + ((u >> 16) & 1u);     // round-to-nearest-even
    return (unsigned short)(u >> 16);
}

// ---------------------------------------------------------------- utilities
__global__ void k_zero(float* __restrict__ p, long n) {
    long i = (long)blockIdx.x * blockDim.x + threadIdx.x;
    if (i < n) p[i] = 0.0f;
}

__global__ void k_deg(const int* __restrict__ dst, float* __restrict__ deg, int E) {
    int e = blockIdx.x * blockDim.x + threadIdx.x;
    if (e < E) unsafeAtomicAdd(&deg[dst[e]], 1.0f);
}

__global__ void k_dis(float* __restrict__ deg, int N) {
    int n = blockIdx.x * blockDim.x + threadIdx.x;
    if (n < N) deg[n] = rsqrtf(deg[n] + 1.0f);   // in-place: deg -> dis
}

// embedding with padding_idx = 0
__global__ void k_embed(const int* __restrict__ tok, const float* __restrict__ emb,
                        float* __restrict__ x, int N) {
    long i = (long)blockIdx.x * blockDim.x + threadIdx.x;
    if (i >= (long)N * DH) return;
    int n = (int)(i >> 7), d = (int)(i & (DH - 1));
    int t = tok[n];
    x[i] = (t == 0) ? 0.0f : emb[(long)t * DH + d];
}

// ---------------------------------------------------------------- WMMA GEMM
// H = X[M,128] @ W[128,128]   (bf16 inputs, f32 accumulate)
// Block: 256 threads / 8 waves; block tile 64x128; wave tile 32x32 (2x2 WMMA).
__global__ __launch_bounds__(256) void k_gemm_wmma(
        const float* __restrict__ X, const float* __restrict__ W,
        float* __restrict__ Hout, int M) {
    __shared__ unsigned short sA[TM][LDP];     // ~17 KB bf16 A tile (padded)
    __shared__ unsigned short sWt[DH][LDP];    // ~35 KB W transposed: sWt[n][k]

    const int t = threadIdx.x;
    const int mbase = blockIdx.x * TM;

    for (int i = t; i < DH * DH; i += 256) {   // stage W (transposed) as bf16
        int k = i >> 7, n = i & (DH - 1);
        sWt[n][k] = f2bf(W[i]);
    }
    for (int i = t; i < TM * DH; i += 256) {   // stage A tile as bf16
        int r = i >> 7, k = i & (DH - 1);
        int row = mbase + r;
        sA[r][k] = f2bf(row < M ? X[(long)row * DH + k] : 0.0f);
    }
    __syncthreads();

    const int wave = t >> 5, lane = t & 31;
    const int wm = (wave >> 2) * 32;           // 0 or 32
    const int wn = (wave & 3) * 32;            // 0,32,64,96
    const int arow = wm + (lane & 15);
    const int hiA = (lane < 16) ? 0 : 8;       // 16-bit A striping (ISA 7.12.2)
    const int hiB = (lane < 16) ? 0 : 8;       // 16-bit B striping
    const int ncol = wn + (lane & 15);

    v8f acc00 = {}, acc01 = {}, acc10 = {}, acc11 = {};
    #pragma unroll
    for (int kb = 0; kb < DH; kb += 32) {
        // A fragments: a[0..7]=K[ko..ko+7], a[8..15]=K[ko+16..ko+23]
        v16bf a0, a1;
        {
            const unsigned short* ap = &sA[arow][kb + hiA];
            v8u lo = *(const v8u*)ap, hi = *(const v8u*)(ap + 16);
            v16u u;
            #pragma unroll
            for (int i = 0; i < 8; ++i) { u[i] = lo[i]; u[8 + i] = hi[i]; }
            a0 = __builtin_bit_cast(v16bf, u);
        }
        {
            const unsigned short* ap = &sA[arow + 16][kb + hiA];
            v8u lo = *(const v8u*)ap, hi = *(const v8u*)(ap + 16);
            v16u u;
            #pragma unroll
            for (int i = 0; i < 8; ++i) { u[i] = lo[i]; u[8 + i] = hi[i]; }
            a1 = __builtin_bit_cast(v16bf, u);
        }
        // B fragments: b[0..7]=K[ko..ko+7], b[8..15]=K[ko+16..ko+23] from transposed W
        v16bf b0, b1;
        {
            const unsigned short* bp = &sWt[ncol][kb + hiB];
            v8u lo = *(const v8u*)bp, hi = *(const v8u*)(bp + 16);
            v16u u;
            #pragma unroll
            for (int i = 0; i < 8; ++i) { u[i] = lo[i]; u[8 + i] = hi[i]; }
            b0 = __builtin_bit_cast(v16bf, u);
        }
        {
            const unsigned short* bp = &sWt[ncol + 16][kb + hiB];
            v8u lo = *(const v8u*)bp, hi = *(const v8u*)(bp + 16);
            v16u u;
            #pragma unroll
            for (int i = 0; i < 8; ++i) { u[i] = lo[i]; u[8 + i] = hi[i]; }
            b1 = __builtin_bit_cast(v16bf, u);
        }
        acc00 = __builtin_amdgcn_wmma_f32_16x16x32_bf16(false, a0, false, b0, (short)0, acc00, false, false);
        acc01 = __builtin_amdgcn_wmma_f32_16x16x32_bf16(false, a0, false, b1, (short)0, acc01, false, false);
        acc10 = __builtin_amdgcn_wmma_f32_16x16x32_bf16(false, a1, false, b0, (short)0, acc10, false, false);
        acc11 = __builtin_amdgcn_wmma_f32_16x16x32_bf16(false, a1, false, b1, (short)0, acc11, false, false);
    }

    // C/D layout: VGPR r -> (lanes 0-15: M=r, N=lane) (lanes 16-31: M=r+8)
    const int srow = (lane < 16) ? 0 : 8;
    const int scol = lane & 15;
    float* p0 = &Hout[(long)(mbase + wm + srow) * DH + wn + scol];
    if (mbase + TM <= M) {                     // interior block: branch-free
        #pragma unroll
        for (int r = 0; r < 8; ++r) {
            float* pr = p0 + (long)r * DH;
            pr[0]            = acc00[r];
            pr[16]           = acc01[r];
            pr[16L * DH]     = acc10[r];
            pr[16L * DH + 16] = acc11[r];
        }
    } else {                                   // tail block: guarded
        #pragma unroll
        for (int r = 0; r < 8; ++r) {
            int m0 = mbase + wm + r + srow;
            if (m0 < M) {
                Hout[(long)m0 * DH + wn + scol]      = acc00[r];
                Hout[(long)m0 * DH + wn + 16 + scol] = acc01[r];
            }
            int m1 = m0 + 16;
            if (m1 < M) {
                Hout[(long)m1 * DH + wn + scol]      = acc10[r];
                Hout[(long)m1 * DH + wn + 16 + scol] = acc11[r];
            }
        }
    }
}

// ---------------------------------------------------------------- edge scatter
// y[dst] += h[src] * dis[src]*dis[dst]   (one wave per edge, float4 per lane)
__global__ void k_scatter(const int* __restrict__ src, const int* __restrict__ dst,
                          const float* __restrict__ dis, const float* __restrict__ h,
                          float* __restrict__ y, int E) {
    long tid = (long)blockIdx.x * blockDim.x + threadIdx.x;
    int e = (int)(tid >> 5), lane = (int)(tid & 31);
    if (e >= E) return;
    int s = src[e], d = dst[e];
    float nm = dis[s] * dis[d];
    float4 v = *(const float4*)&h[(long)s * DH + lane * 4];
    float* yp = &y[(long)d * DH + lane * 4];
    unsafeAtomicAdd(yp + 0, v.x * nm);
    unsafeAtomicAdd(yp + 1, v.y * nm);
    unsafeAtomicAdd(yp + 2, v.z * nm);
    unsafeAtomicAdd(yp + 3, v.w * nm);
}

// self loop + bias + relu:  y = relu(y + h*dis^2 + b)
__global__ void k_finalize(float* __restrict__ y, const float* __restrict__ h,
                           const float* __restrict__ dis, const float* __restrict__ b,
                           int N) {
    long i = (long)blockIdx.x * blockDim.x + threadIdx.x;
    if (i >= (long)N * DH) return;
    int n = (int)(i >> 7), d = (int)(i & (DH - 1));
    float ds = dis[n];
    float v = y[i] + h[i] * ds * ds + b[d];
    y[i] = v > 0.0f ? v : 0.0f;
}

// ---------------------------------------------------------------- pooling
__global__ void k_pool(const int* __restrict__ batch, const float* __restrict__ x,
                       float* __restrict__ pooled, int N) {
    long tid = (long)blockIdx.x * blockDim.x + threadIdx.x;
    int n = (int)(tid >> 5), lane = (int)(tid & 31);
    if (n >= N) return;
    int g = batch[n];
    float4 v = *(const float4*)&x[(long)n * DH + lane * 4];
    float* pp = &pooled[(long)g * DH + lane * 4];
    unsafeAtomicAdd(pp + 0, v.x);
    unsafeAtomicAdd(pp + 1, v.y);
    unsafeAtomicAdd(pp + 2, v.z);
    unsafeAtomicAdd(pp + 3, v.w);
}

__global__ void k_count(const int* __restrict__ batch, float* __restrict__ cnt, int N) {
    int n = blockIdx.x * blockDim.x + threadIdx.x;
    if (n < N) unsafeAtomicAdd(&cnt[batch[n]], 1.0f);
}

// out[g,c] = (pooled[g,:]/max(cnt,1)) @ lin_w + lin_b
__global__ void k_head(const float* __restrict__ pooled, const float* __restrict__ cnt,
                       const float* __restrict__ lw, const float* __restrict__ lb,
                       float* __restrict__ out, int G, int C) {
    int tid = blockIdx.x * blockDim.x + threadIdx.x;
    if (tid >= G * C) return;
    int g = tid / C, c = tid % C;
    float acc = 0.0f;
    #pragma unroll 8
    for (int d = 0; d < DH; ++d) acc += pooled[(long)g * DH + d] * lw[d * C + c];
    float cc = cnt[g];
    out[tid] = acc / (cc > 1.0f ? cc : 1.0f) + lb[c];
}

// ---------------------------------------------------------------- launcher
extern "C" void kernel_launch(void* const* d_in, const int* in_sizes, int n_in,
                              void* d_out, int out_size, void* d_ws, size_t ws_size,
                              hipStream_t stream) {
    const int*   tok   = (const int*)  d_in[0];
    const int*   eidx  = (const int*)  d_in[1];
    const int*   batch = (const int*)  d_in[2];
    const float* emb   = (const float*)d_in[3];
    const float* w1    = (const float*)d_in[4];
    const float* b1    = (const float*)d_in[5];
    const float* w2    = (const float*)d_in[6];
    const float* b2    = (const float*)d_in[7];
    const float* lw    = (const float*)d_in[8];
    const float* lb    = (const float*)d_in[9];

    const int N = in_sizes[0];
    const int E = in_sizes[1] / 2;
    const int C = in_sizes[9];
    const int G = out_size / C;
    const int* src = eidx;          // edge_index[0]
    const int* dst = eidx + E;      // edge_index[1]

    float* ws     = (float*)d_ws;
    float* bufx   = ws;                          // N*128
    float* bufh   = bufx + (long)N * DH;         // N*128
    float* dis    = bufh + (long)N * DH;         // N     (deg -> dis in place)
    float* pooled = dis + N;                     // G*128
    float* cnt    = pooled + (long)G * DH;       // G
    float* out    = (float*)d_out;

    const int B = 256;
    const long NE = (long)N * DH;
    const int gNE   = (int)((NE + B - 1) / B);
    const int gN    = (N + B - 1) / B;
    const int gE    = (E + B - 1) / B;
    const int gE32  = (int)(((long)E * 32 + B - 1) / B);
    const int gN32  = (int)(((long)N * 32 + B - 1) / B);
    const int gGemm = (N + TM - 1) / TM;

    // degrees -> dis
    k_zero <<<gN, B, 0, stream>>>(dis, N);
    k_deg  <<<gE, B, 0, stream>>>(dst, dis, E);
    k_dis  <<<gN, B, 0, stream>>>(dis, N);

    // embedding
    k_embed<<<gNE, B, 0, stream>>>(tok, emb, bufx, N);

    // conv1: h = x@w1 ; x <- relu(scatter(h) + selfloop + b1)
    k_gemm_wmma<<<gGemm, B, 0, stream>>>(bufx, w1, bufh, N);
    k_zero     <<<gNE, B, 0, stream>>>(bufx, NE);
    k_scatter  <<<gE32, B, 0, stream>>>(src, dst, dis, bufh, bufx, E);
    k_finalize <<<gNE, B, 0, stream>>>(bufx, bufh, dis, b1, N);

    // conv2
    k_gemm_wmma<<<gGemm, B, 0, stream>>>(bufx, w2, bufh, N);
    k_zero     <<<gNE, B, 0, stream>>>(bufx, NE);
    k_scatter  <<<gE32, B, 0, stream>>>(src, dst, dis, bufh, bufx, E);
    k_finalize <<<gNE, B, 0, stream>>>(bufx, bufh, dis, b2, N);

    // mean pool + head
    long pz = (long)G * DH + G;
    k_zero <<<(int)((pz + B - 1) / B), B, 0, stream>>>(pooled, pz);
    k_pool <<<gN32, B, 0, stream>>>(batch, bufx, pooled, N);
    k_count<<<gN, B, 0, stream>>>(batch, cnt, N);
    k_head <<<(G * C + B - 1) / B, B, 0, stream>>>(pooled, cnt, lw, lb, out, G, C);
}
